// FineGrainedStudent_20289425506820
// MI455X (gfx1250) — compile-verified, matrix operand order checked by reading, not run on hardware
//
#include <hip/hip_runtime.h>

typedef __attribute__((ext_vector_type(16))) __bf16 v16bf;
typedef __attribute__((ext_vector_type(8)))  float  v8f;
typedef __attribute__((ext_vector_type(4)))  float  v4f;

#define DEV_INLINE __device__ __forceinline__

constexpr int kB  = 32;    // batch
constexpr int kT  = 128;   // frames
constexpr int kR  = 9;     // regions
constexpr int kD  = 512;   // feature dim
constexpr int kB2 = 64;    // 2*B (pos, neg concatenated)
constexpr int CH  = 32;    // K-chunk staged in LDS (one WMMA K step)

// f32 -> bf16 (round to nearest even) and back, branch-free
DEV_INLINE unsigned short f2bf(float v) {
  unsigned u = __float_as_uint(v);
  u += 0x7FFFu + ((u >> 16) & 1u);
  return (unsigned short)(u >> 16);
}
DEV_INLINE float bf2f(unsigned short s) {
  return __uint_as_float(((unsigned)s) << 16);
}

// ---------------------------------------------------------------------------
// Stage 1: sim[bb, i, j] = mean_o max_p  q[b,i,o,:] . t[b,j,p,:]  (masked)
// One workgroup (9 waves, 288 thr) per 16x16 (i,j) tile, per batch, per pair.
// Wave w handles o=w; keeps 9 p-accumulators (16x16 f32 = 8 VGPRs each).
// f32 inputs split into hi/lo bf16 so D = Ahi*Bhi + Ahi*Blo + Alo*Bhi keeps
// ~f32 accuracy while running on the bf16 WMMA pipe.
// ---------------------------------------------------------------------------
__global__ __launch_bounds__(288)
void visil_sim_wmma(const float* __restrict__ q,
                    const float* __restrict__ tpos,
                    const float* __restrict__ tneg,
                    const float* __restrict__ qm,
                    const float* __restrict__ pm,
                    const float* __restrict__ nm,
                    float* __restrict__ sim) {
  const int tile = blockIdx.x;            // 0..63
  const int b    = blockIdx.y;            // 0..31
  const int z    = blockIdx.z;            // 0=pos, 1=neg
  const int i0 = (tile & 7) << 4;
  const int j0 = (tile >> 3) << 4;
  const float* __restrict__ t  = z ? tneg : tpos;
  const float* __restrict__ tm = z ? nm   : pm;

  const int tid  = threadIdx.x;
  const int lane = tid & 31;
  const int o    = tid >> 5;              // wave id == this wave's o, 0..8

  __shared__ unsigned short Ah[9][16][CH];   // q hi-bf16   [o][i][k]
  __shared__ unsigned short Al[9][16][CH];   // q lo-bf16
  __shared__ unsigned short Bh[9][16][CH];   // t hi-bf16   [p][j][k]
  __shared__ unsigned short Bl[9][16][CH];   // t lo-bf16
  __shared__ float red[9][8][32];            // per-o max_p results

  v8f acc[9];
#pragma unroll
  for (int p = 0; p < 9; ++p) { v8f zr = {}; acc[p] = zr; }

  // cooperative loader: 144 rows (9 slots x 16 frames), 2 threads per row
  const int lr  = tid >> 1;               // 0..143
  const int lh  = tid & 1;                // which 16-wide half of the chunk
  const int lo_ = lr / 16;                // o (A) / p (B)
  const int lf_ = lr % 16;                // frame within tile
  const int qrow = ((b * kT + (i0 + lf_)) * kR + lo_) * kD;
  const int trow = ((b * kT + (j0 + lf_)) * kR + lo_) * kD;

  // WMMA 16-bit A/B fragment addressing (ISA 7.12.2):
  // lanes 0-15: elems0-7 = K0..7,  elems8-15 = K16..23
  // lanes16-31: elems0-7 = K8..15, elems8-15 = K24..31
  const int half = lane >> 4;
  const int m    = lane & 15;
  const int off0 = half * 8;
  const int off1 = 16 + half * 8;

  for (int kc = 0; kc < kD; kc += CH) {
    __syncthreads();                      // protect LDS from previous compute
#pragma unroll
    for (int e = 0; e < 16; e += 4) {
      const int k = lh * 16 + e;
      v4f xa = *(const v4f*)(q + qrow + kc + k);
      v4f xb = *(const v4f*)(t + trow + kc + k);
#pragma unroll
      for (int c = 0; c < 4; ++c) {
        unsigned short hsa = f2bf(xa[c]);
        Ah[lo_][lf_][k + c] = hsa;
        Al[lo_][lf_][k + c] = f2bf(xa[c] - bf2f(hsa));
        unsigned short hsb = f2bf(xb[c]);
        Bh[lo_][lf_][k + c] = hsb;
        Bl[lo_][lf_][k + c] = f2bf(xb[c] - bf2f(hsb));
      }
    }
    __syncthreads();

    v16bf ah, al;
#pragma unroll
    for (int e = 0; e < 8; ++e) {
      ah[e]     = __builtin_bit_cast(__bf16, Ah[o][m][off0 + e]);
      ah[8 + e] = __builtin_bit_cast(__bf16, Ah[o][m][off1 + e]);
      al[e]     = __builtin_bit_cast(__bf16, Al[o][m][off0 + e]);
      al[8 + e] = __builtin_bit_cast(__bf16, Al[o][m][off1 + e]);
    }
#pragma unroll
    for (int p = 0; p < 9; ++p) {
      v16bf bh, bl;
#pragma unroll
      for (int e = 0; e < 8; ++e) {
        bh[e]     = __builtin_bit_cast(__bf16, Bh[p][m][off0 + e]);
        bh[8 + e] = __builtin_bit_cast(__bf16, Bh[p][m][off1 + e]);
        bl[e]     = __builtin_bit_cast(__bf16, Bl[p][m][off0 + e]);
        bl[8 + e] = __builtin_bit_cast(__bf16, Bl[p][m][off1 + e]);
      }
      acc[p] = __builtin_amdgcn_wmma_f32_16x16x32_bf16(
          false, ah, false, bh, (short)0, acc[p], false, false);
      acc[p] = __builtin_amdgcn_wmma_f32_16x16x32_bf16(
          false, ah, false, bl, (short)0, acc[p], false, false);
      acc[p] = __builtin_amdgcn_wmma_f32_16x16x32_bf16(
          false, al, false, bh, (short)0, acc[p], false, false);
    }
  }

  // max over p (elementwise on the 16x16 accumulator fragments)
  v8f vmax = acc[0];
#pragma unroll
  for (int p = 1; p < 9; ++p)
#pragma unroll
    for (int r = 0; r < 8; ++r) vmax[r] = fmaxf(vmax[r], acc[p][r]);
#pragma unroll
  for (int r = 0; r < 8; ++r) red[o][r][lane] = vmax[r];
  __syncthreads();

  // mean over o + mask + store.  C layout: VGPR r, lane l -> M = r + 8*(l>=16),
  // N = l & 15.
  if (tid < 256) {
    const int r = tid >> 5, l = tid & 31;
    float s = 0.f;
#pragma unroll
    for (int oo = 0; oo < 9; ++oo) s += red[oo][r][l];
    s *= (1.f / 9.f);
    const int i = i0 + r + ((l >> 4) << 3);
    const int j = j0 + (l & 15);
    const float mv = qm[b * kT + i] * tm[b * kT + j];
    s = (mv > 0.f) ? s : 0.f;
    sim[((z * kB + b) * kT + i) * kT + j] = s;
  }
}

// ---------------------------------------------------------------------------
// Mask after two 2x2 maxpools: m2[bb,y,x] = max over 4x4 block of qm_i*tm_j
// ---------------------------------------------------------------------------
__global__ void visil_mask_pool(const float* __restrict__ qm,
                                const float* __restrict__ pm,
                                const float* __restrict__ nm,
                                float* __restrict__ m2) {
  int idx = blockIdx.x * blockDim.x + threadIdx.x;
  if (idx >= kB2 * 32 * 32) return;
  int x = idx & 31, y = (idx >> 5) & 31, bb = idx >> 10;
  int b = bb & 31, z = bb >> 5;
  const float* tm = z ? nm : pm;
  float v = -__builtin_inff();
  for (int dy = 0; dy < 4; ++dy)
    for (int dx = 0; dx < 4; ++dx)
      v = fmaxf(v, qm[b * kT + 4 * y + dy] * tm[b * kT + 4 * x + dx]);
  m2[idx] = v;
}

// ---------------------------------------------------------------------------
// conv1 (1->32, 3x3, pad1) + relu + maxpool2  : sim(64,1,128,128)->x1(64,32,64,64)
// ---------------------------------------------------------------------------
__global__ void visil_conv1(const float* __restrict__ sim,
                            const float* __restrict__ w1,
                            const float* __restrict__ b1,
                            float* __restrict__ x1) {
  int idx = blockIdx.x * blockDim.x + threadIdx.x;
  if (idx >= kB2 * 32 * 64 * 64) return;
  int x = idx & 63, y = (idx >> 6) & 63, c = (idx >> 12) & 31, bb = idx >> 17;
  float w[9];
#pragma unroll
  for (int k = 0; k < 9; ++k) w[k] = w1[c * 9 + k];
  const float bias = b1[c];
  const float* sp = sim + bb * kT * kT;
  float best = 0.f;   // relu outputs are >= 0, so 0 is a safe identity
#pragma unroll
  for (int dy = 0; dy < 2; ++dy)
#pragma unroll
    for (int dx = 0; dx < 2; ++dx) {
      int Y = 2 * y + dy, X = 2 * x + dx;
      float s = bias;
#pragma unroll
      for (int ky = 0; ky < 3; ++ky) {
        int iy = Y + ky - 1;
        if (iy < 0 || iy >= kT) continue;
#pragma unroll
        for (int kx = 0; kx < 3; ++kx) {
          int ix = X + kx - 1;
          if (ix < 0 || ix >= kT) continue;
          s += w[ky * 3 + kx] * sp[iy * kT + ix];
        }
      }
      best = fmaxf(best, fmaxf(s, 0.f));
    }
  x1[idx] = best;
}

// ---------------------------------------------------------------------------
// conv2 (32->64, 3x3, pad1) + relu + maxpool2 : x1(64,32,64,64)->x2(64,64,32,32)
// One block per (bb, oc); weights staged in LDS; each thread does 4 pixels.
// ---------------------------------------------------------------------------
__global__ __launch_bounds__(256)
void visil_conv2(const float* __restrict__ x1,
                 const float* __restrict__ w2,
                 const float* __restrict__ b2,
                 float* __restrict__ x2) {
  const int bb = blockIdx.x;   // 0..63
  const int oc = blockIdx.y;   // 0..63
  __shared__ float ws[32 * 9];
  for (int t2 = threadIdx.x; t2 < 288; t2 += 256) ws[t2] = w2[oc * 288 + t2];
  __syncthreads();
  const float bias = b2[oc];
  for (int pix = threadIdx.x; pix < 1024; pix += 256) {
    int y = pix >> 5, x = pix & 31;
    float best = 0.f;
    for (int dy = 0; dy < 2; ++dy)
      for (int dx = 0; dx < 2; ++dx) {
        int Y = 2 * y + dy, X = 2 * x + dx;
        float s = bias;
        for (int ic = 0; ic < 32; ++ic) {
          const float* ip = x1 + ((bb * 32 + ic) * 64) * 64;
          const float* wp = ws + ic * 9;
#pragma unroll
          for (int ky = 0; ky < 3; ++ky) {
            int iy = Y + ky - 1;
            if (iy < 0 || iy >= 64) continue;
#pragma unroll
            for (int kx = 0; kx < 3; ++kx) {
              int ix = X + kx - 1;
              if (ix < 0 || ix >= 64) continue;
              s += wp[ky * 3 + kx] * ip[iy * 64 + ix];
            }
          }
        }
        best = fmaxf(best, fmaxf(s, 0.f));
      }
    x2[((bb * 64 + oc) * 32 + y) * 32 + x] = best;
  }
}

// ---------------------------------------------------------------------------
// conv3 (64->128, 3x3, pad1) + relu, fused with final 1x1 conv (128->1) + bias.
// One block (128 threads == oc) per output pixel; input patch staged in LDS.
// ---------------------------------------------------------------------------
__global__ __launch_bounds__(128)
void visil_conv3_final(const float* __restrict__ x2,
                       const float* __restrict__ w3,
                       const float* __restrict__ b3,
                       const float* __restrict__ wf,
                       const float* __restrict__ bf,
                       float* __restrict__ x3) {
  const int bb = blockIdx.x;   // 0..63
  const int y  = blockIdx.y;   // 0..31
  const int x  = blockIdx.z;   // 0..31
  __shared__ float patch[64 * 9];
  __shared__ float rbuf[128];
  for (int t2 = threadIdx.x; t2 < 576; t2 += 128) {
    int ic = t2 / 9, kk = t2 % 9;
    int iy = y + kk / 3 - 1, ix = x + kk % 3 - 1;
    float v = 0.f;
    if (iy >= 0 && iy < 32 && ix >= 0 && ix < 32)
      v = x2[((bb * 64 + ic) * 32 + iy) * 32 + ix];
    patch[t2] = v;
  }
  __syncthreads();
  const int oc = threadIdx.x;
  float s = b3[oc];
  const float* wp = w3 + oc * 576;
  for (int t2 = 0; t2 < 576; ++t2) s += wp[t2] * patch[t2];
  rbuf[oc] = wf[oc] * fmaxf(s, 0.f);
  __syncthreads();
  for (int st = 64; st > 0; st >>= 1) {
    if (oc < st) rbuf[oc] += rbuf[oc + st];
    __syncthreads();
  }
  if (oc == 0) x3[(bb * 32 + y) * 32 + x] = rbuf[0] + bf[0];
}

// ---------------------------------------------------------------------------
// loss + clip + chamfer-v2v.  out[0..63]=v2v (pos then neg), out[64]=loss.
// ---------------------------------------------------------------------------
__global__ void visil_zero(float* out) { if (threadIdx.x == 0) out[64] = 0.f; }

__global__ __launch_bounds__(32)
void visil_finalize(const float* __restrict__ x3,
                    const float* __restrict__ m2,
                    float* __restrict__ out) {
  const int bb = blockIdx.x;   // 0..63
  const int i  = threadIdx.x;  // row 0..31
  float lossp = 0.f, rowmax = -__builtin_inff(), mm = -__builtin_inff();
  for (int j = 0; j < 32; ++j) {
    float s = x3[(bb * 32 + i) * 32 + j];
    float m = m2[(bb * 32 + i) * 32 + j];
    lossp += fmaxf(-1.f - s, 0.f) + fmaxf(s - 1.f, 0.f);
    float sc = fminf(fmaxf(s, -1.f), 1.f);
    rowmax = fmaxf(rowmax, (m > 0.f) ? sc : -__builtin_inff());
    mm = fmaxf(mm, m);
  }
  float rowval = (mm > 0.f) ? rowmax : 0.f;
  __shared__ float rs[32], rm[32], rl[32];
  rs[i] = rowval; rm[i] = mm; rl[i] = lossp;
  __syncthreads();
  if (i == 0) {
    float num = 0.f, den = 0.f, lt = 0.f;
    for (int k2 = 0; k2 < 32; ++k2) { num += rs[k2]; den += rm[k2]; lt += rl[k2]; }
    out[bb] = num / den;
    atomicAdd(out + 64, lt);
  }
}

// ---------------------------------------------------------------------------
extern "C" void kernel_launch(void* const* d_in, const int* in_sizes, int n_in,
                              void* d_out, int out_size, void* d_ws, size_t ws_size,
                              hipStream_t stream) {
  (void)in_sizes; (void)n_in; (void)out_size; (void)ws_size;
  const float* q  = (const float*)d_in[0];   // anchors   (B,T,R,D)
  const float* tp = (const float*)d_in[1];   // positives
  const float* tn = (const float*)d_in[2];   // negatives
  const float* qm = (const float*)d_in[3];   // anchors_masks   (B,T)
  const float* pm = (const float*)d_in[4];   // positive_masks
  const float* nm = (const float*)d_in[5];   // negative_masks
  const float* w1 = (const float*)d_in[6];
  const float* b1 = (const float*)d_in[7];
  const float* w2 = (const float*)d_in[8];
  const float* b2 = (const float*)d_in[9];
  const float* w3 = (const float*)d_in[10];
  const float* b3 = (const float*)d_in[11];
  const float* wf = (const float*)d_in[12];
  const float* bf = (const float*)d_in[13];
  float* out = (float*)d_out;

  // workspace layout (f32), total ~53 MB
  float* ws  = (float*)d_ws;
  float* sim = ws;                               // 64*128*128
  float* x1  = sim + 64 * 128 * 128;             // 64*32*64*64
  float* x2  = x1 + 64 * 32 * 64 * 64;           // 64*64*32*32
  float* x3  = x2 + 64 * 64 * 32 * 32;           // 64*32*32
  float* m2  = x3 + 64 * 32 * 32;                // 64*32*32

  visil_sim_wmma<<<dim3(64, 32, 2), 288, 0, stream>>>(q, tp, tn, qm, pm, nm, sim);
  visil_mask_pool<<<(kB2 * 32 * 32 + 255) / 256, 256, 0, stream>>>(qm, pm, nm, m2);
  visil_conv1<<<(kB2 * 32 * 64 * 64 + 255) / 256, 256, 0, stream>>>(sim, w1, b1, x1);
  visil_conv2<<<dim3(64, 64), 256, 0, stream>>>(x1, w2, b2, x2);
  visil_conv3_final<<<dim3(64, 32, 32), 128, 0, stream>>>(x2, w3, b3, wf, bf, x3);
  visil_zero<<<1, 1, 0, stream>>>(out);
  visil_finalize<<<64, 32, 0, stream>>>(x3, m2, out);
}
